// SourceModel_34248069219248
// MI455X (gfx1250) — compile-verified, matrix operand order checked by reading, not run on hardware
//
#include <hip/hip_runtime.h>
#include <cstdint>

// ---------------------------------------------------------------------------
// GRU LM on MI455X (gfx1250, wave32):
//  - all GEMMs via v_wmma_f32_16x16x32_bf16 (f32 accumulate)
//  - output/input GEMMs: WG tile 128x128, wave tile 32x64; B operand staged
//    through LDS (double-buffered) via GLOBAL_LOAD_ASYNC_TO_LDS_B128
//    (ASYNCcnt + s_wait_asynccnt), sync ds stores as fallback
//  - recurrence: 256 stream-ordered step kernels; each wave owns 16 H-columns
//    across all 3 gates so gate math + state update is wave-local
// ---------------------------------------------------------------------------

#define DEV __device__ __forceinline__

typedef __attribute__((ext_vector_type(16))) __bf16 v16bf;
typedef __attribute__((ext_vector_type(8)))  float  v8f;
typedef __attribute__((ext_vector_type(4)))  int    v4i;

#if defined(__has_builtin)
#if __has_builtin(__builtin_amdgcn_global_load_async_to_lds_b128) && \
    __has_builtin(__builtin_amdgcn_s_wait_asynccnt)
#define HAVE_ASYNC_LDS 1
#endif
#endif
#ifndef HAVE_ASYNC_LDS
#define HAVE_ASYNC_LDS 0
#endif

#if HAVE_ASYNC_LDS
// Builtin expects: (AS1 v4i* src, AS3 v4i* dst, imm offset, imm cpol)
typedef __attribute__((address_space(1))) v4i* gas_v4i;
typedef __attribute__((address_space(3))) v4i* las_v4i;
#endif

static constexpr int Bsz = 16, Tlen = 256, Edim = 512, Hdim = 1024, Vdim = 32000;

DEV unsigned short f2bf(float f) {
  uint32_t u = __float_as_uint(f);
  uint32_t r = u + 0x7FFFu + ((u >> 16) & 1u);   // round-to-nearest-even
  return (unsigned short)(r >> 16);
}

// A/B fragment for V_WMMA_F32_16X16X32_BF16 (CDNA5 ISA 7.12.2):
// lane holds row (lane&15); K values = {8*(lane>>4)+0..7} U {16+8*(lane>>4)+0..7}
// => two contiguous 16-byte runs per lane.  Works for global or LDS pointers.
DEV v16bf load_frag(const unsigned short* rowk, int half) {
  union { v16bf v; uint4 q[2]; } u;
  u.q[0] = *reinterpret_cast<const uint4*>(rowk + half * 8);
  u.q[1] = *reinterpret_cast<const uint4*>(rowk + 16 + half * 8);
  return u.v;
}

DEV v8f wmma_bf16(v16bf a, v16bf b, v8f c) {
  return __builtin_amdgcn_wmma_f32_16x16x32_bf16(false, a, false, b, (short)0, c,
                                                 false, false);
}

// ------------------------------ utility kernels -----------------------------

__global__ void k_f32_to_bf16(const float* __restrict__ src,
                              unsigned short* __restrict__ dst, int n) {
  for (int i = blockIdx.x * blockDim.x + threadIdx.x; i < n;
       i += gridDim.x * blockDim.x)
    dst[i] = f2bf(src[i]);
}

__global__ void k_embed(const int* __restrict__ x, const float* __restrict__ emb,
                        unsigned short* __restrict__ xe, int n) {
  for (int i = blockIdx.x * blockDim.x + threadIdx.x; i < n;
       i += gridDim.x * blockDim.x) {
    int bt = i >> 9;              // Edim = 512
    int e  = i & (Edim - 1);
    int tok = x[bt];
    xe[i] = f2bf(emb[(size_t)tok * Edim + e]);
  }
}

// h0 = tanh(z @ W_init^T + b_init): 16M MACs, plain fp32 is fine.
__global__ void k_h0(const float* __restrict__ z, const float* __restrict__ Winit,
                     const float* __restrict__ binit, float* __restrict__ hbuf0,
                     unsigned short* __restrict__ hs) {
  int idx = blockIdx.x * blockDim.x + threadIdx.x;   // 16 * 1024
  int b = idx >> 10, j = idx & (Hdim - 1);
  const float* zr = z + (size_t)b * Hdim;
  const float* wr = Winit + (size_t)j * Hdim;
  float s = 0.f;
  for (int k = 0; k < Hdim; k += 4)
    s += zr[k] * wr[k] + zr[k + 1] * wr[k + 1] + zr[k + 2] * wr[k + 2] +
         zr[k + 3] * wr[k + 3];
  float h = tanhf(s + binit[j]);
  hbuf0[(size_t)b * Hdim + j] = h;
  hs[((size_t)b * Tlen) * Hdim + j] = f2bf(h);   // hs[:,0,:] = h0
}

// ------------------------- generic bf16 WMMA GEMM ---------------------------
// C[M,N] = A[M,K](bf16,row-major) * B[N,K](bf16,row-major)^T + bias[N]  (fp32 out)
// WG: 256 thr = 8 waves arranged 4(M) x 2(N); wave tile 32x64 -> WG tile 128x128.
// B block (128 rows x 32 k = 8KB) staged through LDS, double-buffered; async
// global->LDS so the copy overlaps the current K-step's WMMAs.
// Requires M%128==0, N%128==0, K%32==0.
__global__ void __launch_bounds__(256) k_wmma_gemm_bias(
    const unsigned short* __restrict__ A, int lda,
    const unsigned short* __restrict__ Bm, int ldb,
    const float* __restrict__ bias,
    float* __restrict__ C, int ldc, int K) {
  constexpr int WGN = 128, KB = 32;
  __shared__ alignas(16) unsigned short ldsB[2][WGN * KB];   // 2 x 8 KB

  const int tid  = threadIdx.x;
  const int lane = tid & 31;
  const int wave = tid >> 5;
  const int wm = wave & 3;            // 4 waves in M
  const int wn = wave >> 2;           // 2 waves in N
  const int rowW  = blockIdx.y * 128 + wm * 32;
  const int colWG = blockIdx.x * WGN;
  const int colW  = wn * 64;          // within WG tile
  const int half = lane >> 4, nl = lane & 15;

  const unsigned short* a0 = A + (size_t)(rowW + nl) * lda;
  const unsigned short* a1 = A + (size_t)(rowW + 16 + nl) * lda;

  // B staging: 128 rows x 32 elems = 512 x 16B chunks, 2 per thread.
  const int c0 = tid * 2, c1 = c0 + 1;
  const int r0 = c0 >> 2, s0 = c0 & 3;
  const int r1 = c1 >> 2, s1 = c1 & 3;
  const unsigned short* g0 = Bm + (size_t)(colWG + r0) * ldb + s0 * 8;
  const unsigned short* g1 = Bm + (size_t)(colWG + r1) * ldb + s1 * 8;
  const int l0 = r0 * KB + s0 * 8;
  const int l1 = r1 * KB + s1 * 8;

  auto stage = [&](int buf, int k0) {
#if HAVE_ASYNC_LDS
    __builtin_amdgcn_global_load_async_to_lds_b128(
        (gas_v4i)(g0 + k0), (las_v4i)&ldsB[buf][l0], 0, 0);
    __builtin_amdgcn_global_load_async_to_lds_b128(
        (gas_v4i)(g1 + k0), (las_v4i)&ldsB[buf][l1], 0, 0);
#else
    *(uint4*)&ldsB[buf][l0] = *(const uint4*)(g0 + k0);
    *(uint4*)&ldsB[buf][l1] = *(const uint4*)(g1 + k0);
#endif
  };

  v8f acc[2][4] = {};
  stage(0, 0);
#if HAVE_ASYNC_LDS
  __builtin_amdgcn_s_wait_asynccnt(0);
#endif
  __syncthreads();

  int buf = 0;
  for (int k0 = 0; k0 < K; k0 += KB) {
    if (k0 + KB < K) stage(buf ^ 1, k0 + KB);   // overlap next block copy
    __builtin_prefetch(a0 + k0 + 2 * KB, 0, 0); // global_prefetch_b8
    __builtin_prefetch(a1 + k0 + 2 * KB, 0, 0);
    v16bf fa0 = load_frag(a0 + k0, half);
    v16bf fa1 = load_frag(a1 + k0, half);
#pragma unroll
    for (int j = 0; j < 4; ++j) {
      const unsigned short* brow = &ldsB[buf][(colW + j * 16 + nl) * KB];
      v16bf fb = load_frag(brow, half);          // ds_load_b128 x2
      acc[0][j] = wmma_bf16(fa0, fb, acc[0][j]);
      acc[1][j] = wmma_bf16(fa1, fb, acc[1][j]);
    }
#if HAVE_ASYNC_LDS
    __builtin_amdgcn_s_wait_asynccnt(0);
#endif
    __syncthreads();
    buf ^= 1;
  }

#pragma unroll
  for (int j = 0; j < 4; ++j) {
    int col = colWG + colW + j * 16 + nl;
    float bv = bias[col];
#pragma unroll
    for (int r = 0; r < 8; ++r) {
      int rw = rowW + r + 8 * half;
      C[(size_t)rw * ldc + col]        = acc[0][j][r] + bv;
      C[(size_t)(rw + 16) * ldc + col] = acc[1][j][r] + bv;
    }
  }
}

// ------------------------------- GRU step -----------------------------------
// One launch per timestep t.  Grid = 16 WGs x 128 thr (4 waves).
// Wave owns columns j in [jbase, jbase+16): computes gh for r/z/n gates
// (3 WMMA tiles, K=1024), then the state update — fully wave-local.
// A operand = hs_bf16[:, t, :] (written by the previous step / h0 kernel).
__global__ void __launch_bounds__(128) k_gru_step(
    const unsigned short* __restrict__ hsA, const unsigned short* __restrict__ Whh,
    const float* __restrict__ gi, const float* __restrict__ bhh,
    const float* __restrict__ hin, float* __restrict__ hout,
    unsigned short* __restrict__ hs_out, int t) {
  int lane = threadIdx.x & 31;
  int wave = threadIdx.x >> 5;
  int jbase = blockIdx.x * 64 + wave * 16;
  int half = lane >> 4, nl = lane & 15;

  const unsigned short* arow = hsA + ((size_t)nl * Tlen + t) * Hdim;  // batch nl
  const unsigned short* brR = Whh + (size_t)(jbase + nl) * Hdim;
  const unsigned short* brZ = Whh + (size_t)(Hdim + jbase + nl) * Hdim;
  const unsigned short* brN = Whh + (size_t)(2 * Hdim + jbase + nl) * Hdim;

  v8f aR = {}, aZ = {}, aN = {};
  for (int k0 = 0; k0 < Hdim; k0 += 32) {
    v16bf fa = load_frag(arow + k0, half);
    aR = wmma_bf16(fa, load_frag(brR + k0, half), aR);
    aZ = wmma_bf16(fa, load_frag(brZ + k0, half), aZ);
    aN = wmma_bf16(fa, load_frag(brN + k0, half), aN);
  }

  int j = jbase + nl;
  float bR = bhh[j], bZ = bhh[Hdim + j], bN = bhh[2 * Hdim + j];
#pragma unroll
  for (int r = 0; r < 8; ++r) {
    int bidx = r + 8 * half;   // batch index (C-layout row)
    size_t gbase = ((size_t)bidx * Tlen + t) * (3 * Hdim);
    float ir = gi[gbase + j];
    float iz = gi[gbase + Hdim + j];
    float in = gi[gbase + 2 * Hdim + j];
    float ghr = aR[r] + bR, ghz = aZ[r] + bZ, ghn = aN[r] + bN;
    float rg = 1.f / (1.f + __expf(-(ir + ghr)));
    float zg = 1.f / (1.f + __expf(-(iz + ghz)));
    float ng = tanhf(in + rg * ghn);
    float hold = hin[(size_t)bidx * Hdim + j];
    float hnew = (1.f - zg) * ng + zg * hold;
    hout[(size_t)bidx * Hdim + j] = hnew;
    if (t + 1 < Tlen)
      hs_out[((size_t)bidx * Tlen + (t + 1)) * Hdim + j] = f2bf(hnew);
  }
}

// ------------------------------- launcher -----------------------------------

extern "C" void kernel_launch(void* const* d_in, const int* in_sizes, int n_in,
                              void* d_out, int out_size, void* d_ws, size_t ws_size,
                              hipStream_t stream) {
  const float* z     = (const float*)d_in[0];
  const int*   x     = (const int*)  d_in[1];
  const float* emb   = (const float*)d_in[2];
  const float* Winit = (const float*)d_in[3];
  const float* binit = (const float*)d_in[4];
  const float* Wih   = (const float*)d_in[5];
  const float* Whh   = (const float*)d_in[6];
  const float* bih   = (const float*)d_in[7];
  const float* bhh   = (const float*)d_in[8];
  const float* Wout  = (const float*)d_in[9];
  const float* bout  = (const float*)d_in[10];
  float* out = (float*)d_out;

  char* ws = (char*)d_ws;
  size_t off = 0;
  auto alloc = [&](size_t bytes) {
    size_t o = off;
    off = (off + bytes + 255) & ~size_t(255);
    return o;
  };
  unsigned short* xe      = (unsigned short*)(ws + alloc((size_t)Bsz * Tlen * Edim * 2));
  unsigned short* wih_bf  = (unsigned short*)(ws + alloc((size_t)3 * Hdim * Edim * 2));
  unsigned short* whh_bf  = (unsigned short*)(ws + alloc((size_t)3 * Hdim * Hdim * 2));
  unsigned short* wout_bf = (unsigned short*)(ws + alloc((size_t)Vdim * Hdim * 2));
  unsigned short* hs      = (unsigned short*)(ws + alloc((size_t)Bsz * Tlen * Hdim * 2));
  float*          gi      = (float*)(ws + alloc((size_t)Bsz * Tlen * 3 * Hdim * 4));
  float*          hbuf    = (float*)(ws + alloc((size_t)2 * Bsz * Hdim * 4));

  const int thr = 256;
  k_f32_to_bf16<<<512,  thr, 0, stream>>>(Wih,  wih_bf,  3 * Hdim * Edim);
  k_f32_to_bf16<<<1024, thr, 0, stream>>>(Whh,  whh_bf,  3 * Hdim * Hdim);
  k_f32_to_bf16<<<4096, thr, 0, stream>>>(Wout, wout_bf, Vdim * Hdim);
  k_embed<<<2048, thr, 0, stream>>>(x, emb, xe, Bsz * Tlen * Edim);
  k_h0<<<(Bsz * Hdim) / thr, thr, 0, stream>>>(z, Winit, binit, hbuf, hs);

  // gi_all = xe @ W_ih^T + b_ih : [4096,3072], K=512
  dim3 g1(3 * Hdim / 128, (Bsz * Tlen) / 128);
  k_wmma_gemm_bias<<<g1, 256, 0, stream>>>(xe, Edim, wih_bf, Edim, bih, gi,
                                           3 * Hdim, Edim);

  // sequential recurrence
  for (int t = 0; t < Tlen; ++t) {
    const float* hin = hbuf + (size_t)(t & 1) * Bsz * Hdim;
    float*      hout = hbuf + (size_t)((t + 1) & 1) * Bsz * Hdim;
    k_gru_step<<<Hdim / 64, 128, 0, stream>>>(hs, whh_bf, gi, bhh, hin, hout,
                                              hs, t);
  }

  // logits = hs @ W_out^T + b_out : [4096,32000], K=1024
  dim3 g2(Vdim / 128, (Bsz * Tlen) / 128);
  k_wmma_gemm_bias<<<g2, 256, 0, stream>>>(hs, Hdim, wout_bf, Hdim, bout, out,
                                           Vdim, Hdim);
}
